// ArcFaceLoss_55817394979146
// MI455X (gfx1250) — matrix operand order repaired and checked by reading, not compiled, not run on hardware
//
#include <hip/hip_runtime.h>
#include <hip/hip_bf16.h>
#include <math.h>

// ---------------------------------------------------------------------------
// ArcFace loss, MI455X (gfx1250, wave32).
//   x: [512,512] f32, w: [100000,512] f32, target: [512] i32 -> scalar f32.
// Pipeline:
//   K1 xnorm : row-normalize x, emit f32 + bf16 hi/lo split
//   K2 wnorm : 1/||w_row||  (one wave32 per row)
//   K3 gemm  : 128x128 tiles, bf16x3 WMMA (hi*hi + hi*lo + lo*hi),
//              software-pipelined global->reg->LDS, fused ArcFace margin +
//              online-softmax partials (logits never materialized)
//   K4 lse   : merge (max,sumexp) partials -> logsumexp per row
//   K5 tgt   : target-column logit + per-row NLL
//   K6 mean  : scalar mean
// Grid order for K3: 4 row-blocks of one column block dispatched
// consecutively so the shared 128-col weight tile stays in L2 (192MB).
// ---------------------------------------------------------------------------

typedef __attribute__((ext_vector_type(16))) __bf16 v16bf;
typedef __attribute__((ext_vector_type(8)))  __bf16 v8bf;
typedef __attribute__((ext_vector_type(8)))  float  v8f;

#define BATCH 512
#define DIM   512
#define NCLS  100000
#define NBN   782            // ceil(100000/128) column blocks
#define NPB   (NBN * 2)      // partials per row (2 n-waves per block)
#define SCALE 30.0f
#define COS_M 0.8775825618903728f
#define SIN_M 0.4794255386042030f

__device__ __forceinline__ void ls_merge(float& m, float& s, float om, float os) {
    float nm = fmaxf(m, om);
    float ns = s * __expf(m - nm) + os * __expf(om - nm);
    ns = (nm == -INFINITY) ? 0.0f : ns;   // branchless guard
    m = nm; s = ns;
}

// ---------------- K1: normalize x rows, produce f32 + bf16 hi/lo -----------
__global__ __launch_bounds__(256)
void k_xnorm(const float* __restrict__ x, float* __restrict__ xn,
             __bf16* __restrict__ xnhi, __bf16* __restrict__ xnlo) {
    __shared__ float red[256];
    const int row = blockIdx.x, tid = threadIdx.x;
    float a0 = x[row * DIM + tid];
    float a1 = x[row * DIM + tid + 256];
    red[tid] = a0 * a0 + a1 * a1;
    __syncthreads();
    for (int off = 128; off > 0; off >>= 1) {
        if (tid < off) red[tid] += red[tid + off];
        __syncthreads();
    }
    const float inv = 1.0f / fmaxf(sqrtf(red[0]), 1e-12f);
    for (int j = 0; j < 2; ++j) {
        int k = tid + j * 256;
        float v = x[row * DIM + k] * inv;
        xn[row * DIM + k] = v;
        __bf16 h = (__bf16)v;
        xnhi[row * DIM + k] = h;
        xnlo[row * DIM + k] = (__bf16)(v - (float)h);
    }
}

// ---------------- K2: 1/||w_row||, one wave32 per row ----------------------
__global__ __launch_bounds__(256)
void k_wnorm(const float* __restrict__ w, float* __restrict__ winv) {
    const int lane = threadIdx.x & 31;
    const int row  = blockIdx.x * 8 + (threadIdx.x >> 5);
    float ss = 0.0f;
    for (int k = lane; k < DIM; k += 32) {
        float v = w[(size_t)row * DIM + k];
        ss += v * v;
    }
    for (int m = 16; m >= 1; m >>= 1) ss += __shfl_xor(ss, m);
    if (lane == 0) winv[row] = 1.0f / fmaxf(sqrtf(ss), 1e-12f);
}

// ---------------- K3: WMMA GEMM + fused margin + softmax partials ----------
__global__ __launch_bounds__(256)
void k_gemm(const __bf16* __restrict__ xnhi, const __bf16* __restrict__ xnlo,
            const float* __restrict__ w, const float* __restrict__ winv,
            const int* __restrict__ target, float2* __restrict__ partials) {
    // LDS tiles, padded stride 40 elements (80B) to spread banks, 16B aligned.
    __shared__ __align__(16) __bf16 sAhi[128 * 40];
    __shared__ __align__(16) __bf16 sAlo[128 * 40];
    __shared__ __align__(16) __bf16 sBhi[128 * 40];
    __shared__ __align__(16) __bf16 sBlo[128 * 40];

    const int tid  = threadIdx.x;
    const int lane = tid & 31;
    const int wave = tid >> 5;
    const int wm   = wave & 3;    // 4 M-waves: rows  wm*32 .. +31 (2 tiles)
    const int wn   = wave >> 2;   // 2 N-waves: cols  wn*64 .. +63 (4 tiles)
    const int m0   = blockIdx.x * 128;   // row block (4 total, fast axis)
    const int n0   = blockIdx.y * 128;   // col block (782 total, slow axis)

    // global->LDS mapping: 128 rows, 2 threads/row, 16 elems each
    const int am = tid >> 1, ak = (tid & 1) * 16;    // A rows / k-part
    const int bc = tid >> 1, bk = (tid & 1) * 16;    // B rows / k-part

    const int   cG = n0 + bc;
    const int   cc = cG < NCLS ? cG : NCLS - 1;
    const float iv = winv[cc];

    const __bf16* aph = xnhi + (size_t)(m0 + am) * DIM + ak;
    const __bf16* apl = xnlo + (size_t)(m0 + am) * DIM + ak;
    const float*  wpb = w + (size_t)cc * DIM + bk;

    v8f acc[2][4];
    #pragma unroll
    for (int mt = 0; mt < 2; ++mt)
        #pragma unroll
        for (int t = 0; t < 4; ++t) acc[mt][t] = (v8f){0,0,0,0,0,0,0,0};

    const int mr = lane & 15;
    const int hf = lane >> 4;

    // ---- software pipeline: prefetch K-tile 0 into registers
    v8bf rah0 = *(const v8bf*)(aph + 0), rah1 = *(const v8bf*)(aph + 8);
    v8bf ral0 = *(const v8bf*)(apl + 0), ral1 = *(const v8bf*)(apl + 8);
    float4 rw0 = *(const float4*)(wpb + 0),  rw1 = *(const float4*)(wpb + 4);
    float4 rw2 = *(const float4*)(wpb + 8),  rw3 = *(const float4*)(wpb + 12);

    for (int kt = 0; kt < DIM / 32; ++kt) {
        // ---- stage registers -> LDS
        *(v8bf*)&sAhi[am * 40 + ak]     = rah0;
        *(v8bf*)&sAhi[am * 40 + ak + 8] = rah1;
        *(v8bf*)&sAlo[am * 40 + ak]     = ral0;
        *(v8bf*)&sAlo[am * 40 + ak + 8] = ral1;
        {
            float f[16] = {rw0.x*iv, rw0.y*iv, rw0.z*iv, rw0.w*iv,
                           rw1.x*iv, rw1.y*iv, rw1.z*iv, rw1.w*iv,
                           rw2.x*iv, rw2.y*iv, rw2.z*iv, rw2.w*iv,
                           rw3.x*iv, rw3.y*iv, rw3.z*iv, rw3.w*iv};
            v8bf h0, h1, l0, l1;
            #pragma unroll
            for (int e = 0; e < 16; ++e) {
                __bf16 h = (__bf16)f[e];
                __bf16 l = (__bf16)(f[e] - (float)h);
                if (e < 8) { h0[e] = h; l0[e] = l; }
                else       { h1[e - 8] = h; l1[e - 8] = l; }
            }
            *(v8bf*)&sBhi[bc * 40 + bk]     = h0;
            *(v8bf*)&sBhi[bc * 40 + bk + 8] = h1;
            *(v8bf*)&sBlo[bc * 40 + bk]     = l0;
            *(v8bf*)&sBlo[bc * 40 + bk + 8] = l1;
        }
        __syncthreads();

        // ---- issue next K-tile's global loads (overlap with WMMA below)
        if (kt < DIM / 32 - 1) {
            const int kn = (kt + 1) * 32;
            rah0 = *(const v8bf*)(aph + kn);     rah1 = *(const v8bf*)(aph + kn + 8);
            ral0 = *(const v8bf*)(apl + kn);     ral1 = *(const v8bf*)(apl + kn + 8);
            rw0  = *(const float4*)(wpb + kn);   rw1  = *(const float4*)(wpb + kn + 4);
            rw2  = *(const float4*)(wpb + kn + 8); rw3 = *(const float4*)(wpb + kn + 12);
        }

        // ---- A fragments per ISA 16-bit A 16x32 layout:
        // lanes<16: K 0..7 (v0-3), 16..23 (v4-7); lanes>=16: K 8..15, 24..31
        v16bf ah[2], al[2];
        #pragma unroll
        for (int mt = 0; mt < 2; ++mt) {
            const __bf16* p = &sAhi[(wm * 32 + mt * 16 + mr) * 40 + hf * 8];
            v8bf x0 = *(const v8bf*)p, x1 = *(const v8bf*)(p + 16);
            ah[mt] = __builtin_shufflevector(x0, x1,
                       0,1,2,3,4,5,6,7,8,9,10,11,12,13,14,15);
            const __bf16* q = &sAlo[(wm * 32 + mt * 16 + mr) * 40 + hf * 8];
            v8bf y0 = *(const v8bf*)q, y1 = *(const v8bf*)(q + 16);
            al[mt] = __builtin_shufflevector(y0, y1,
                       0,1,2,3,4,5,6,7,8,9,10,11,12,13,14,15);
        }

        #pragma unroll
        for (int t = 0; t < 4; ++t) {
            // B fragment: lanes<16 K 0..7, 16..23; lanes>=16 K 8..15, 24..31, N = lane&15
            const __bf16* bp = &sBhi[(wn * 64 + t * 16 + mr) * 40 + hf * 8];
            v8bf b0 = *(const v8bf*)bp, b1 = *(const v8bf*)(bp + 16);
            v16bf bh = __builtin_shufflevector(b0, b1,
                         0,1,2,3,4,5,6,7,8,9,10,11,12,13,14,15);
            const __bf16* bq = &sBlo[(wn * 64 + t * 16 + mr) * 40 + hf * 8];
            v8bf c0 = *(const v8bf*)bq, c1 = *(const v8bf*)(bq + 16);
            v16bf bl = __builtin_shufflevector(c0, c1,
                         0,1,2,3,4,5,6,7,8,9,10,11,12,13,14,15);

            #pragma unroll
            for (int mt = 0; mt < 2; ++mt) {
                acc[mt][t] = __builtin_amdgcn_wmma_f32_16x16x32_bf16(
                    false, ah[mt], false, bh, (short)0, acc[mt][t], false, false);
                acc[mt][t] = __builtin_amdgcn_wmma_f32_16x16x32_bf16(
                    false, ah[mt], false, bl, (short)0, acc[mt][t], false, false);
                acc[mt][t] = __builtin_amdgcn_wmma_f32_16x16x32_bf16(
                    false, al[mt], false, bh, (short)0, acc[mt][t], false, false);
            }
        }
        __syncthreads();
    }

    // ---- Epilogue: branchless ArcFace margin + online softmax (64 cols/wave)
    // D layout: VGPR v -> M = v + 8*hf, N = lane&15 within each 16x16 tile.
    const int pcol = blockIdx.y * 2 + wn;
    #pragma unroll
    for (int mt = 0; mt < 2; ++mt) {
        #pragma unroll
        for (int v = 0; v < 8; ++v) {
            const int row = m0 + wm * 32 + mt * 16 + v + 8 * hf;
            const int tg  = target[row];
            float l[4];
            #pragma unroll
            for (int t = 0; t < 4; ++t) {
                const int col = n0 + wn * 64 + t * 16 + mr;
                float c   = acc[mt][t][v];
                float sn  = sqrtf(fmaxf(1.0f - c * c, 0.0f));
                float phi = c * COS_M - sn * SIN_M;
                float pm  = (c > 0.0f) ? phi : c;          // easy margin
                float lg  = (col == tg) ? SCALE * pm : SCALE * c;
                l[t] = (col < NCLS) ? lg : -INFINITY;
            }
            float mx = fmaxf(fmaxf(l[0], l[1]), fmaxf(l[2], l[3]));
            float sm = __expf(l[0] - mx) + __expf(l[1] - mx) +
                       __expf(l[2] - mx) + __expf(l[3] - mx);
            sm = (mx == -INFINITY) ? 0.0f : sm;
            // butterfly across the 16 lanes of this half-wave
            for (int msk = 1; msk <= 8; msk <<= 1) {
                float om = __shfl_xor(mx, msk);
                float os = __shfl_xor(sm, msk);
                ls_merge(mx, sm, om, os);
            }
            if (mr == 0) partials[(size_t)row * NPB + pcol] = make_float2(mx, sm);
        }
    }
}

// ---------------- K4: per-row logsumexp from partials ----------------------
__global__ __launch_bounds__(256)
void k_lse(const float2* __restrict__ partials, float* __restrict__ lse) {
    __shared__ float sm_[256], ss_[256];
    const int row = blockIdx.x, tid = threadIdx.x;
    float m = -INFINITY, s = 0.0f;
    for (int i = tid; i < NPB; i += 256) {
        float2 p = partials[(size_t)row * NPB + i];
        ls_merge(m, s, p.x, p.y);
    }
    sm_[tid] = m; ss_[tid] = s;
    __syncthreads();
    for (int off = 128; off > 0; off >>= 1) {
        if (tid < off) {
            float mm = sm_[tid], st = ss_[tid];
            ls_merge(mm, st, sm_[tid + off], ss_[tid + off]);
            sm_[tid] = mm; ss_[tid] = st;
        }
        __syncthreads();
    }
    if (tid == 0) lse[row] = sm_[0] + logf(ss_[0]);
}

// ---------------- K5: target logit + per-row NLL ---------------------------
__global__ __launch_bounds__(256)
void k_tgt(const float* __restrict__ xn, const float* __restrict__ w,
           const float* __restrict__ winv, const int* __restrict__ target,
           const float* __restrict__ lse, float* __restrict__ lossrow) {
    __shared__ float red[256];
    const int row = blockIdx.x, tid = threadIdx.x;
    const int tg = target[row];
    float d = 0.0f;
    for (int k = tid; k < DIM; k += 256)
        d += xn[row * DIM + k] * w[(size_t)tg * DIM + k];
    red[tid] = d;
    __syncthreads();
    for (int off = 128; off > 0; off >>= 1) {
        if (tid < off) red[tid] += red[tid + off];
        __syncthreads();
    }
    if (tid == 0) {
        float c   = red[0] * winv[tg];
        float sn  = sqrtf(fmaxf(1.0f - c * c, 0.0f));
        float phi = c * COS_M - sn * SIN_M;
        float lt  = SCALE * ((c > 0.0f) ? phi : c);
        lossrow[row] = lse[row] - lt;
    }
}

// ---------------- K6: mean --------------------------------------------------
__global__ __launch_bounds__(256)
void k_mean(const float* __restrict__ lossrow, float* __restrict__ out) {
    __shared__ float red[256];
    const int tid = threadIdx.x;
    red[tid] = lossrow[tid] + lossrow[tid + 256];
    __syncthreads();
    for (int off = 128; off > 0; off >>= 1) {
        if (tid < off) red[tid] += red[tid + off];
        __syncthreads();
    }
    if (tid == 0) out[0] = red[0] / (float)BATCH;
}

// ---------------------------------------------------------------------------
extern "C" void kernel_launch(void* const* d_in, const int* in_sizes, int n_in,
                              void* d_out, int out_size, void* d_ws, size_t ws_size,
                              hipStream_t stream) {
    const float* x  = (const float*)d_in[0];
    const float* w  = (const float*)d_in[1];
    const int*   tg = (const int*)d_in[2];

    char* ws = (char*)d_ws;
    size_t o = 0;
    float*  xn      = (float*)(ws + o);  o += (size_t)BATCH * DIM * 4;      // 1.0 MB
    __bf16* xnhi    = (__bf16*)(ws + o); o += (size_t)BATCH * DIM * 2;      // 0.5 MB
    __bf16* xnlo    = (__bf16*)(ws + o); o += (size_t)BATCH * DIM * 2;      // 0.5 MB
    float*  winv    = (float*)(ws + o);  o += (size_t)NCLS * 4;             // 0.4 MB
    float2* parts   = (float2*)(ws + o); o += (size_t)BATCH * NPB * 8;      // 6.4 MB
    float*  lse     = (float*)(ws + o);  o += (size_t)BATCH * 4;
    float*  lossrow = (float*)(ws + o);  o += (size_t)BATCH * 4;

    k_xnorm<<<BATCH, 256, 0, stream>>>(x, xn, xnhi, xnlo);
    k_wnorm<<<NCLS / 8, 256, 0, stream>>>(w, winv);
    // x = 4 row-blocks (fast), y = 782 col-blocks (slow): L2 reuse of w tiles
    k_gemm<<<dim3(BATCH / 128, NBN), 256, 0, stream>>>(xnhi, xnlo, w, winv, tg, parts);
    k_lse<<<BATCH, 256, 0, stream>>>(parts, lse);
    k_tgt<<<BATCH, 256, 0, stream>>>(xn, w, winv, tg, lse, lossrow);
    k_mean<<<1, 256, 0, stream>>>(lossrow, (float*)d_out);
}
